// SciBert_BiLSTM_CRF_59365037965660
// MI455X (gfx1250) — compile-verified, matrix operand order checked by reading, not run on hardware
//
#include <hip/hip_runtime.h>
#include <hip/hip_bf16.h>

#define SEQ  8192
#define EDIM 768
#define GDIM 2048   // 4 * H2
#define HDIM 512    // H2
#define NTAG 5
#define START_TAG 3
#define STOP_TAG  4
#define NEGINF   -10000.0f

typedef __attribute__((ext_vector_type(16))) __bf16         bf16x16;
typedef __attribute__((ext_vector_type(16))) unsigned short ushort16;
typedef __attribute__((ext_vector_type(8)))  unsigned short ushort8;
typedef __attribute__((ext_vector_type(8)))  float          floatx8;

__device__ __forceinline__ unsigned short f32_to_bf16_rne(float f) {
  unsigned int u = __builtin_bit_cast(unsigned int, f);
  u += 0x7FFFu + ((u >> 16) & 1u);   // round-to-nearest-even
  return (unsigned short)(u >> 16);
}

__device__ __forceinline__ float sigmoidf_fast(float x) {
  return 1.0f / (1.0f + __expf(-x));
}

// ---------------------------------------------------------------------------
// Kernel 0: one-time fp32 -> bf16 conversion (8 elements / thread, b128 I/O).
// Amortized across 512 (weights) / 128 (embeddings) reuses in the GEMM.
// ---------------------------------------------------------------------------
__global__ __launch_bounds__(256) void cvt_f32_bf16(
    const float* __restrict__ src, unsigned int* __restrict__ dst, int n8) {
  const int i = blockIdx.x * blockDim.x + threadIdx.x;
  if (i >= n8) return;
  const float* s = src + (size_t)i * 8;
  const float4 a = *(const float4*)(s);
  const float4 b = *(const float4*)(s + 4);
  uint4 o;
  o.x = (unsigned)f32_to_bf16_rne(a.x) | ((unsigned)f32_to_bf16_rne(a.y) << 16);
  o.y = (unsigned)f32_to_bf16_rne(a.z) | ((unsigned)f32_to_bf16_rne(a.w) << 16);
  o.z = (unsigned)f32_to_bf16_rne(b.x) | ((unsigned)f32_to_bf16_rne(b.y) << 16);
  o.w = (unsigned)f32_to_bf16_rne(b.z) | ((unsigned)f32_to_bf16_rne(b.w) << 16);
  *(uint4*)(dst + (size_t)i * 4) = o;
}

// ---------------------------------------------------------------------------
// Kernel 1: xg[m,n] = emb[m,:] @ w_ih[n,:] + b_ih[n] + b_hh[n]
// bf16 WMMA 16x16x32. Each wave owns TWO 16x16 C tiles along N sharing one
// A fragment; 8 waves -> 64x64 of C per block. Operands pre-converted bf16.
// ---------------------------------------------------------------------------
__global__ __launch_bounds__(256) void xg_gemm_wmma(
    const unsigned short* __restrict__ a16,   // [SEQ,  EDIM] bf16
    const unsigned short* __restrict__ w16,   // [GDIM, EDIM] bf16
    const float* __restrict__ b_ih, const float* __restrict__ b_hh,
    float* __restrict__ xg) {
  const int lane = threadIdx.x & 31;
  const int wave = threadIdx.x >> 5;                  // 0..7
  const int m0 = blockIdx.y * 64 + (wave & 3) * 16;   // C tile row base
  const int n0 = blockIdx.x * 64 + (wave >> 2) * 32;  // C tile col base (2 tiles)
  const int r  = lane & 15;
  const int kb = (lane >> 4) * 8;                     // K interleave base

  const unsigned short* arow  = a16 + (size_t)(m0 + r) * EDIM + kb;
  const unsigned short* brow0 = w16 + (size_t)(n0 + r) * EDIM + kb;
  const unsigned short* brow1 = w16 + (size_t)(n0 + 16 + r) * EDIM + kb;

  union Frag { ushort16 v; ushort8 h[2]; };
  floatx8 acc0 = {}, acc1 = {};
#pragma unroll 4
  for (int k = 0; k < EDIM; k += 32) {
    Frag fa, f0, f1;
    // per-lane K pattern: {kb..kb+7, kb+16..kb+23} within the 32-wide K tile
    fa.h[0] = *(const ushort8*)(arow + k);
    fa.h[1] = *(const ushort8*)(arow + k + 16);
    f0.h[0] = *(const ushort8*)(brow0 + k);
    f0.h[1] = *(const ushort8*)(brow0 + k + 16);
    f1.h[0] = *(const ushort8*)(brow1 + k);
    f1.h[1] = *(const ushort8*)(brow1 + k + 16);
    const bf16x16 av = __builtin_bit_cast(bf16x16, fa.v);
    acc0 = __builtin_amdgcn_wmma_f32_16x16x32_bf16(
        false, av, false, __builtin_bit_cast(bf16x16, f0.v), (short)0, acc0, false, false);
    acc1 = __builtin_amdgcn_wmma_f32_16x16x32_bf16(
        false, av, false, __builtin_bit_cast(bf16x16, f1.v), (short)0, acc1, false, false);
  }
  // C layout: VGPR v -> row m = m0 + v + 8*(lane>>4), col n = n0 + (lane&15)
  const int na = n0 + (lane & 15);
  const int nb = na + 16;
  const float bias_a = b_ih[na] + b_hh[na];
  const float bias_b = b_ih[nb] + b_hh[nb];
#pragma unroll
  for (int v = 0; v < 8; ++v) {
    const int m = m0 + v + 8 * (lane >> 4);
    xg[(size_t)m * GDIM + na] = acc0[v] + bias_a;
    xg[(size_t)m * GDIM + nb] = acc1[v] + bias_b;
  }
}

// ---------------------------------------------------------------------------
// Kernel 2: sequential LSTM scan, one persistent block per direction.
// h, c, gates live in LDS; w_hh streamed from L2 each step.
// ---------------------------------------------------------------------------
__global__ __launch_bounds__(1024) void lstm_scan(
    const float* __restrict__ xg_f, const float* __restrict__ xg_b,
    const float* __restrict__ w_hh_f, const float* __restrict__ w_hh_b,
    const float* __restrict__ h0_f, const float* __restrict__ c0_f,
    const float* __restrict__ h0_b, const float* __restrict__ c0_b,
    float* __restrict__ lstm_out) {
  const int dir = blockIdx.x;  // 0 = forward, 1 = backward
  const float* xg   = dir ? xg_b : xg_f;
  const float* w_hh = dir ? w_hh_b : w_hh_f;
  const float* h0   = dir ? h0_b : h0_f;
  const float* c0   = dir ? c0_b : c0_f;

  __shared__ __align__(16) float sh[HDIM];
  __shared__ __align__(16) float sc[HDIM];
  __shared__ __align__(16) float sg[GDIM];

  const int j = threadIdx.x;  // 0..1023
  if (j < HDIM) { sh[j] = h0[j]; sc[j] = c0[j]; }
  __syncthreads();

  for (int s = 0; s < SEQ; ++s) {
    const int t = dir ? (SEQ - 1 - s) : s;
    const float* xr = xg + (size_t)t * GDIM;
    if (s + 1 < SEQ) {  // pull next timestep's gate row toward L0/L2
      const int tn = dir ? (t - 1) : (t + 1);
      __builtin_prefetch(xg + (size_t)tn * GDIM + (size_t)j * 2, 0, 0);
    }
    // each thread produces gates j and j+1024: g[n] = xr[n] + w_hh[n,:] . h
#pragma unroll
    for (int p = 0; p < 2; ++p) {
      const int n = j + p * 1024;
      const float* wr = w_hh + (size_t)n * HDIM;
      float acc = xr[n];
#pragma unroll 8
      for (int k = 0; k < HDIM; k += 4) {
        const float4 w4 = *(const float4*)(wr + k);
        const float4 h4 = *(const float4*)(sh + k);
        acc = fmaf(w4.x, h4.x, acc);
        acc = fmaf(w4.y, h4.y, acc);
        acc = fmaf(w4.z, h4.z, acc);
        acc = fmaf(w4.w, h4.w, acc);
      }
      sg[n] = acc;
    }
    __syncthreads();
    if (j < HDIM) {  // PyTorch gate order: i, f, g, o in blocks of HDIM
      const float ig = sigmoidf_fast(sg[j]);
      const float fg = sigmoidf_fast(sg[HDIM + j]);
      const float gg = tanhf(sg[2 * HDIM + j]);
      const float og = sigmoidf_fast(sg[3 * HDIM + j]);
      const float c  = fmaf(fg, sc[j], ig * gg);
      const float h  = og * tanhf(c);
      sc[j] = c;
      sh[j] = h;
      lstm_out[(size_t)t * (2 * HDIM) + dir * HDIM + j] = h;
    }
    __syncthreads();
  }
}

// ---------------------------------------------------------------------------
// Kernel 3: feats = lstm_out @ w_tag.T + b_tag   ([S,1024] x [1024,5])
// ---------------------------------------------------------------------------
__global__ __launch_bounds__(256) void tag_proj(
    const float* __restrict__ lstm_out, const float* __restrict__ w_tag,
    const float* __restrict__ b_tag, float* __restrict__ feats) {
  const int id = blockIdx.x * blockDim.x + threadIdx.x;
  if (id >= SEQ * NTAG) return;
  const int row = id / NTAG, tg = id % NTAG;
  const float* x = lstm_out + (size_t)row * (2 * HDIM);
  const float* w = w_tag + (size_t)tg * (2 * HDIM);
  float acc = b_tag[tg];
#pragma unroll 8
  for (int k = 0; k < 2 * HDIM; k += 4) {
    const float4 a = *(const float4*)(x + k);
    const float4 b = *(const float4*)(w + k);
    acc = fmaf(a.x, b.x, fmaf(a.y, b.y, fmaf(a.z, b.z, fmaf(a.w, b.w, acc))));
  }
  feats[id] = acc;
}

// ---------------------------------------------------------------------------
// Kernel 4: Viterbi forward scan + backtrack (T=5, trivially serial).
// out[0] = path_score, out[1..S] = best path tags (as float).
// ---------------------------------------------------------------------------
__global__ void viterbi_kernel(const float* __restrict__ feats,
                               const float* __restrict__ trans,
                               float* __restrict__ out,
                               int* __restrict__ bp) {
  if (threadIdx.x != 0 || blockIdx.x != 0) return;
  float tr[NTAG * NTAG];
#pragma unroll
  for (int i = 0; i < NTAG * NTAG; ++i) tr[i] = trans[i];
  float fv[NTAG];
#pragma unroll
  for (int i = 0; i < NTAG; ++i) fv[i] = (i == START_TAG) ? 0.0f : NEGINF;

  for (int t = 0; t < SEQ; ++t) {
    float nf[NTAG];
#pragma unroll
    for (int nx = 0; nx < NTAG; ++nx) {
      float best = -3.4e38f;
      int bi = 0;
#pragma unroll
      for (int pv = 0; pv < NTAG; ++pv) {  // first-max wins (matches argmax)
        const float m = fv[pv] + tr[nx * NTAG + pv];
        if (m > best) { best = m; bi = pv; }
      }
      nf[nx] = best + feats[t * NTAG + nx];
      bp[t * NTAG + nx] = bi;
    }
#pragma unroll
    for (int nx = 0; nx < NTAG; ++nx) fv[nx] = nf[nx];
  }
  float best = -3.4e38f;
  int bt = 0;
#pragma unroll
  for (int i = 0; i < NTAG; ++i) {
    const float v = fv[i] + tr[STOP_TAG * NTAG + i];
    if (v > best) { best = v; bt = i; }
  }
  out[0] = best;
  int tag = bt;
  for (int t = SEQ - 1; t >= 0; --t) {
    out[1 + t] = (float)tag;
    tag = bp[t * NTAG + tag];
  }
}

// ---------------------------------------------------------------------------
extern "C" void kernel_launch(void* const* d_in, const int* in_sizes, int n_in,
                              void* d_out, int out_size, void* d_ws, size_t ws_size,
                              hipStream_t stream) {
  const float* emb    = (const float*)d_in[0];
  const float* w_ih_f = (const float*)d_in[1];
  const float* w_hh_f = (const float*)d_in[2];
  const float* b_ih_f = (const float*)d_in[3];
  const float* b_hh_f = (const float*)d_in[4];
  const float* w_ih_b = (const float*)d_in[5];
  const float* w_hh_b = (const float*)d_in[6];
  const float* b_ih_b = (const float*)d_in[7];
  const float* b_hh_b = (const float*)d_in[8];
  const float* w_tag  = (const float*)d_in[9];
  const float* b_tag  = (const float*)d_in[10];
  const float* h0_f   = (const float*)d_in[11];
  const float* c0_f   = (const float*)d_in[12];
  const float* h0_b   = (const float*)d_in[13];
  const float* c0_b   = (const float*)d_in[14];
  const float* trans  = (const float*)d_in[15];
  (void)in_sizes; (void)n_in; (void)out_size; (void)ws_size;

  float* ws    = (float*)d_ws;
  float* xg_f  = ws;                                 // [SEQ, GDIM] f32
  float* xg_b  = xg_f + (size_t)SEQ * GDIM;          // [SEQ, GDIM] f32
  float* lstm  = xg_b + (size_t)SEQ * GDIM;          // [SEQ, 2*HDIM] f32
  float* feats = lstm + (size_t)SEQ * (2 * HDIM);    // [SEQ, NTAG] f32
  int*   bp    = (int*)(feats + (size_t)SEQ * NTAG); // [SEQ, NTAG] i32
  unsigned short* emb16 = (unsigned short*)(bp + (size_t)SEQ * NTAG); // [SEQ, EDIM] bf16
  unsigned short* w16_f = emb16 + (size_t)SEQ * EDIM;                 // [GDIM, EDIM] bf16
  unsigned short* w16_b = w16_f + (size_t)GDIM * EDIM;                // [GDIM, EDIM] bf16

  // one-time bf16 pre-conversion of GEMM operands
  const int n8_emb = SEQ * EDIM / 8, n8_w = GDIM * EDIM / 8;
  cvt_f32_bf16<<<(n8_emb + 255) / 256, 256, 0, stream>>>(emb, (unsigned int*)emb16, n8_emb);
  cvt_f32_bf16<<<(n8_w + 255) / 256, 256, 0, stream>>>(w_ih_f, (unsigned int*)w16_f, n8_w);
  cvt_f32_bf16<<<(n8_w + 255) / 256, 256, 0, stream>>>(w_ih_b, (unsigned int*)w16_b, n8_w);

  dim3 gemm_grid(GDIM / 64, SEQ / 64);
  xg_gemm_wmma<<<gemm_grid, 256, 0, stream>>>(emb16, w16_f, b_ih_f, b_hh_f, xg_f);
  xg_gemm_wmma<<<gemm_grid, 256, 0, stream>>>(emb16, w16_b, b_ih_b, b_hh_b, xg_b);
  lstm_scan<<<2, 1024, 0, stream>>>(xg_f, xg_b, w_hh_f, w_hh_b,
                                    h0_f, c0_f, h0_b, c0_b, lstm);
  tag_proj<<<(SEQ * NTAG + 255) / 256, 256, 0, stream>>>(lstm, w_tag, b_tag, feats);
  viterbi_kernel<<<1, 32, 0, stream>>>(feats, trans, (float*)d_out, bp);
}
